// TreeModel_90134183674494
// MI455X (gfx1250) — matrix operand order repaired and checked by reading, not compiled, not run on hardware
//
#include <hip/hip_runtime.h>
#include <math.h>

// ---------------------------------------------------------------------------
// Types for CDNA5 WMMA (wave32).  v_wmma_f32_16x16x32_bf16: D = A(16x32) x
// B(32x16) + C(16x16 f32).
// ---------------------------------------------------------------------------
typedef __bf16 v16bf __attribute__((ext_vector_type(16)));
typedef __bf16 v8bf  __attribute__((ext_vector_type(8)));
typedef float  v8f   __attribute__((ext_vector_type(8)));

#define B_SZ    4096
#define D_IN    512
#define D_EMB   1024
#define C_OUT   128
#define N_INNER 15
#define N_LEAF  16
#define BETA    10.0f

// Scheduler fence: nothing may be reordered across this point.  Used between
// a bounded load group and its consuming WMMAs; loads of the NEXT group can
// still overlap the current WMMA group (no fence between them).
#if defined(__has_builtin)
#if __has_builtin(__builtin_amdgcn_sched_barrier)
#define SCHED_FENCE() __builtin_amdgcn_sched_barrier(0)
#endif
#endif
#ifndef SCHED_FENCE
#define SCHED_FENCE() asm volatile("" ::: "memory")
#endif

__device__ __forceinline__ v8f wmma_bf16(v16bf a, v16bf b, v8f c) {
    return __builtin_amdgcn_wmma_f32_16x16x32_bf16(
        /*neg_a=*/false, a, /*neg_b=*/false, b,
        /*c_mod=*/(short)0, c, /*reuse_a=*/false, /*reuse_b=*/false);
}

// Build a 16-element bf16 fragment from two contiguous 16-byte runs.
__device__ __forceinline__ v16bf frag2(const __bf16* p0, const __bf16* p1) {
    v8bf lo = *(const v8bf*)p0;
    v8bf hi = *(const v8bf*)p1;
    return __builtin_shufflevector(lo, hi, 0,1,2,3,4,5,6,7,8,9,10,11,12,13,14,15);
}

// A fragment at byte-constant offset from a per-lane base (base already
// includes the +8h K-shift): runs [ks*32, +8) and [ks*32+16, +8) elements.
__device__ __forceinline__ v16bf fragA(const __bf16* abase, int ks) {
    const __bf16* p = abase + ks * 32;
    return frag2(p, p + 16);
}

// B fragment: 16 contiguous bf16 per lane (two b128 loads).
__device__ __forceinline__ v16bf fragB(const __bf16* p) {
    return frag2(p, p + 8);
}

// ---------------------------------------------------------------------------
// K0a: f32 -> bf16 elementwise (x)
// ---------------------------------------------------------------------------
__global__ void k_cvt_bf16(const float* __restrict__ src,
                           __bf16* __restrict__ dst, int n) {
    int i = blockIdx.x * 256 + threadIdx.x;
    if (i < n) dst[i] = (__bf16)src[i];
}

// ---------------------------------------------------------------------------
// K0b: swizzle f32 [mat][K][N] weights into WMMA-B fragment-major bf16:
//   dst[mat][ntile][kstep][lane][e] = src[mat][kstep*32 + 8*(lane>>4) + e + (e>=8 ? 8 : 0)]
//                                        [ntile*16 + (lane&15)]
// so the GEMM loop fetches each B fragment as two contiguous b128 loads at
// immediate offsets from one per-lane base pointer.
// ---------------------------------------------------------------------------
__global__ void k_swizzle(const float* __restrict__ src,
                          __bf16* __restrict__ dst,
                          int N, int KS, int NT,
                          size_t sstride, size_t dstride, int total) {
    int i = blockIdx.x * 256 + threadIdx.x;
    if (i >= total) return;
    int e     = i & 15;
    int lane  = (i >> 4) & 31;
    int rest  = i >> 9;
    int kstep = rest % KS;  rest /= KS;
    int ntile = rest % NT;
    int mat   = rest / NT;
    int h = lane >> 4, n = lane & 15;
    int row = kstep * 32 + 8 * h + e + (e >= 8 ? 8 : 0);
    int col = ntile * 16 + n;
    dst[(size_t)mat * dstride +
        ((((size_t)ntile * KS + kstep) * 32 + lane) << 4) + e] =
        (__bf16)src[(size_t)mat * sstride + (size_t)row * N + col];
}

// ---------------------------------------------------------------------------
// K1: emb = relu(x @ W_enc + b_enc), stored bf16 row-major [4096][1024].
// 256 threads = 8 waves; WG = one 16-row b-tile x all 1024 cols
// (8 n-tiles per wave). K = 512 -> 16 k-steps.
// Fenced groups of 4 B fragments: group g+1's loads overlap group g's WMMAs.
// ---------------------------------------------------------------------------
__global__ void k1_encoder(const __bf16* __restrict__ xbf,
                           const __bf16* __restrict__ wsw,
                           const float* __restrict__ b_enc,
                           __bf16* __restrict__ emb) {
    const int tid = threadIdx.x;
    const int wave = tid >> 5, lane = tid & 31;
    const int m16 = lane & 15, h = lane >> 4;
    const int b0 = blockIdx.x * 16;
    const int nt0 = wave * 8;

    v8f acc[8];
#pragma unroll
    for (int t = 0; t < 8; ++t)
#pragma unroll
        for (int r = 0; r < 8; ++r) acc[t][r] = 0.0f;

    const __bf16* abase = xbf + (size_t)(b0 + m16) * D_IN + 8 * h;
    const __bf16* bbase = wsw + ((size_t)nt0 * 16 * 32 + lane) * 16;

#pragma unroll 2
    for (int ks = 0; ks < 16; ++ks) {
        const __bf16* bk = bbase + ks * 512;        // +1024 bytes / step
        v16bf a = fragA(abase, ks);
#pragma unroll
        for (int g = 0; g < 8; g += 4) {
            v16bf bf0 = fragB(bk + (g + 0) * 8192);
            v16bf bf1 = fragB(bk + (g + 1) * 8192);
            v16bf bf2 = fragB(bk + (g + 2) * 8192);
            v16bf bf3 = fragB(bk + (g + 3) * 8192);
            SCHED_FENCE();
            acc[g + 0] = wmma_bf16(a, bf0, acc[g + 0]);
            acc[g + 1] = wmma_bf16(a, bf1, acc[g + 1]);
            acc[g + 2] = wmma_bf16(a, bf2, acc[g + 2]);
            acc[g + 3] = wmma_bf16(a, bf3, acc[g + 3]);
        }
    }

#pragma unroll
    for (int t = 0; t < 8; ++t) {
        int col = (nt0 + t) * 16 + m16;
        float bias = b_enc[col];
#pragma unroll
        for (int r = 0; r < 8; ++r) {
            float v = acc[t][r] + bias;
            v = v > 0.0f ? v : 0.0f;
            emb[(size_t)(b0 + r + 8 * h) * D_EMB + col] = (__bf16)v;
        }
    }
}

// ---------------------------------------------------------------------------
// K2: fused inner-node GEMM + argmax + sigmoid routing (86% of all FLOPs).
// Grid (128 b-tiles of 32 rows, 15 nodes), 256 threads = 8 waves.
// Each wave: 8 n-tiles x 2 row-blocks (M=32): each B fragment feeds two
// WMMAs (~27 FLOP/byte from L2).  Fenced groups of 2 B fragments keep
// register pressure bounded (acc=128) while overlapping next group's loads
// with the current 4 WMMAs.
// Waves 0-3 cover cols [0,512) (the "w" half, tracked for argmax);
// waves 4-7 cover cols [512,1024) (the "b" half, parked in LDS as bf16).
// ---------------------------------------------------------------------------
__global__ void k2_inner(const __bf16* __restrict__ embbf,
                         const __bf16* __restrict__ wsw,
                         const float* __restrict__ b_inner,
                         const float* __restrict__ x,
                         float* __restrict__ p_right) {
    __shared__ __bf16 biasLDS[32][512];  // 32 KB: wb columns 512..1023
    __shared__ float  candV[4][32];
    __shared__ int    candI[4][32];

    const int tid = threadIdx.x;
    const int wave = tid >> 5, lane = tid & 31;
    const int m16 = lane & 15, h = lane >> 4;
    const int b0 = blockIdx.x * 32;
    const int node = blockIdx.y;
    const int nt0 = wave * 8;

    const __bf16* abase0 = embbf + (size_t)(b0 + m16) * D_EMB + 8 * h;
    const __bf16* abase1 = abase0 + (size_t)16 * D_EMB;
    const __bf16* bbase = wsw + (size_t)node * (64 * 32 * 32 * 16) +
                          ((size_t)nt0 * 32 * 32 + lane) * 16;

    v8f acc0[8], acc1[8];
#pragma unroll
    for (int t = 0; t < 8; ++t)
#pragma unroll
        for (int r = 0; r < 8; ++r) { acc0[t][r] = 0.0f; acc1[t][r] = 0.0f; }

    for (int ks = 0; ks < 32; ++ks) {
        const __bf16* bk = bbase + ks * 512;        // +1024 bytes / step
        v16bf a0 = fragA(abase0, ks);
        v16bf a1 = fragA(abase1, ks);
#pragma unroll
        for (int g = 0; g < 8; g += 2) {
            v16bf bf0 = fragB(bk + (g + 0) * 16384);
            v16bf bf1 = fragB(bk + (g + 1) * 16384);
            SCHED_FENCE();
            acc0[g + 0] = wmma_bf16(a0, bf0, acc0[g + 0]);
            acc1[g + 0] = wmma_bf16(a1, bf0, acc1[g + 0]);
            acc0[g + 1] = wmma_bf16(a0, bf1, acc0[g + 1]);
            acc1[g + 1] = wmma_bf16(a1, bf1, acc1[g + 1]);
        }
    }

    const float* bvec = b_inner + (size_t)node * (2 * D_IN);

    if (wave < 4) {
        // "w" half: per-row running (max, argmax) across this wave's columns.
        float mv0[8], mv1[8];
        int   mi0[8], mi1[8];
#pragma unroll
        for (int r = 0; r < 8; ++r) {
            mv0[r] = -3.4e38f; mi0[r] = 0;
            mv1[r] = -3.4e38f; mi1[r] = 0;
        }
#pragma unroll
        for (int t = 0; t < 8; ++t) {
            int col = (nt0 + t) * 16 + m16;         // in [0, 512)
            float bias = bvec[col];
#pragma unroll
            for (int r = 0; r < 8; ++r) {
                float v0 = acc0[t][r] + bias;
                if (v0 > mv0[r]) { mv0[r] = v0; mi0[r] = col; }
                float v1 = acc1[t][r] + bias;
                if (v1 > mv1[r]) { mv1[r] = v1; mi1[r] = col; }
            }
        }
        // reduce across the 16 lanes of each half-wave (rows match within half)
#pragma unroll
        for (int off = 1; off < 16; off <<= 1) {
#pragma unroll
            for (int r = 0; r < 8; ++r) {
                float ov0 = __shfl_xor(mv0[r], off);
                int   oi0 = __shfl_xor(mi0[r], off);
                if (ov0 > mv0[r]) { mv0[r] = ov0; mi0[r] = oi0; }
                float ov1 = __shfl_xor(mv1[r], off);
                int   oi1 = __shfl_xor(mi1[r], off);
                if (ov1 > mv1[r]) { mv1[r] = ov1; mi1[r] = oi1; }
            }
        }
        if (m16 == 0) {
#pragma unroll
            for (int r = 0; r < 8; ++r) {
                candV[wave][r + 8 * h]      = mv0[r];
                candI[wave][r + 8 * h]      = mi0[r];
                candV[wave][16 + r + 8 * h] = mv1[r];
                candI[wave][16 + r + 8 * h] = mi1[r];
            }
        }
    } else {
        // "b" half: stash wb[:,512:1024] in LDS for the selected-index lookup.
#pragma unroll
        for (int t = 0; t < 8; ++t) {
            int col = (nt0 + t) * 16 + m16;         // in [512, 1024)
            float bias = bvec[col];
#pragma unroll
            for (int r = 0; r < 8; ++r) {
                biasLDS[r + 8 * h][col - 512]      = (__bf16)(acc0[t][r] + bias);
                biasLDS[16 + r + 8 * h][col - 512] = (__bf16)(acc1[t][r] + bias);
            }
        }
    }
    __syncthreads();

    if (tid < 32) {
        int m = tid;
        float mv = candV[0][m];
        int   mi = candI[0][m];
#pragma unroll
        for (int w = 1; w < 4; ++w) {
            float v = candV[w][m];
            if (v > mv) { mv = v; mi = candI[w][m]; }
        }
        float bb = (float)biasLDS[m][mi];            // wb[m][mi + 512]
        float xv = x[(size_t)(b0 + m) * D_IN + mi];
        float sel = BETA * (mv * xv + bb);
        float p = 1.0f / (1.0f + __expf(-sel));
        p_right[(size_t)(b0 + m) * N_INNER + node] = p;
    }
}

// ---------------------------------------------------------------------------
// K3: fused leaf GEMMs + tree-probability weighting + combine.
// 256 threads = 8 waves, 2 leaves per wave, LDS ds_add_f32 reduction.
// Fenced groups of 4 B fragments, as in K1.
// ---------------------------------------------------------------------------
__global__ void k3_leaf(const __bf16* __restrict__ embbf,
                        const __bf16* __restrict__ wsw,
                        const float* __restrict__ b_leaf,
                        const float* __restrict__ p_right,
                        float* __restrict__ out) {
    __shared__ float probs[16][16];
    __shared__ float outl[16][128];

    const int tid = threadIdx.x;
    const int wave = tid >> 5, lane = tid & 31;
    const int m16 = lane & 15, h = lane >> 4;
    const int b0 = blockIdx.x * 16;

    for (int i = tid; i < 16 * 128; i += 256) ((float*)outl)[i] = 0.0f;

    if (tid < 16) {
        int m = tid;
        const float* pr = p_right + (size_t)(b0 + m) * N_INNER;
#pragma unroll
        for (int leaf = 0; leaf < 16; ++leaf) {
            float prob = 1.0f;
            int idx = 0, off = 0;
#pragma unroll
            for (int level = 0; level < 4; ++level) {
                int bit = (leaf >> (3 - level)) & 1;
                float p = pr[off + idx];
                prob *= bit ? p : (1.0f - p);
                off += (1 << level);
                idx = idx * 2 + bit;
            }
            probs[m][leaf] = prob;
        }
    }
    __syncthreads();

    const __bf16* abase = embbf + (size_t)(b0 + m16) * D_EMB + 8 * h;

    for (int leaf = wave; leaf < N_LEAF; leaf += 8) {
        const __bf16* bbase = wsw + (size_t)leaf * (8 * 32 * 32 * 16) +
                              (size_t)lane * 16;
        v8f acc[8];
#pragma unroll
        for (int t = 0; t < 8; ++t)
#pragma unroll
            for (int r = 0; r < 8; ++r) acc[t][r] = 0.0f;

#pragma unroll 2
        for (int ks = 0; ks < 32; ++ks) {
            const __bf16* bk = bbase + ks * 512;    // +1024 bytes / step
            v16bf a = fragA(abase, ks);
#pragma unroll
            for (int g = 0; g < 8; g += 4) {
                v16bf bf0 = fragB(bk + (g + 0) * 16384);
                v16bf bf1 = fragB(bk + (g + 1) * 16384);
                v16bf bf2 = fragB(bk + (g + 2) * 16384);
                v16bf bf3 = fragB(bk + (g + 3) * 16384);
                SCHED_FENCE();
                acc[g + 0] = wmma_bf16(a, bf0, acc[g + 0]);
                acc[g + 1] = wmma_bf16(a, bf1, acc[g + 1]);
                acc[g + 2] = wmma_bf16(a, bf2, acc[g + 2]);
                acc[g + 3] = wmma_bf16(a, bf3, acc[g + 3]);
            }
        }

#pragma unroll
        for (int t = 0; t < 8; ++t) {
            int col = t * 16 + m16;
            float bl = b_leaf[(size_t)leaf * C_OUT + col];
#pragma unroll
            for (int r = 0; r < 8; ++r) {
                int m = r + 8 * h;
                float v = (acc[t][r] + bl) * probs[m][leaf];
                atomicAdd(&outl[m][col], v);
            }
        }
    }
    __syncthreads();

    for (int i = tid; i < 16 * 128; i += 256) {
        int m = i >> 7, c = i & 127;
        out[(size_t)(b0 + m) * C_OUT + c] = outl[m][c];
    }
}

// ---------------------------------------------------------------------------
// Host launcher
// ---------------------------------------------------------------------------
extern "C" void kernel_launch(void* const* d_in, const int* in_sizes, int n_in,
                              void* d_out, int out_size, void* d_ws, size_t ws_size,
                              hipStream_t stream) {
    const float* x       = (const float*)d_in[0];
    const float* W_enc   = (const float*)d_in[1];
    const float* b_enc   = (const float*)d_in[2];
    const float* W_inner = (const float*)d_in[3];
    const float* b_inner = (const float*)d_in[4];
    const float* W_leaf  = (const float*)d_in[5];
    const float* b_leaf  = (const float*)d_in[6];
    float* out = (float*)d_out;

    char* ws = (char*)d_ws;
    __bf16* xbf       = (__bf16*)ws;  ws += (size_t)B_SZ * D_IN * 2;
    __bf16* wenc_sw   = (__bf16*)ws;  ws += (size_t)64 * 16 * 32 * 16 * 2;
    __bf16* winner_sw = (__bf16*)ws;  ws += (size_t)N_INNER * 64 * 32 * 32 * 16 * 2;
    __bf16* wleaf_sw  = (__bf16*)ws;  ws += (size_t)N_LEAF * 8 * 32 * 32 * 16 * 2;
    __bf16* emb       = (__bf16*)ws;  ws += (size_t)B_SZ * D_EMB * 2;
    float*  p_right   = (float*)ws;   ws += (size_t)B_SZ * N_INNER * 4;

    // Pre-pass: bf16 conversion + fragment-major weight swizzles.
    {
        int n = B_SZ * D_IN;
        k_cvt_bf16<<<(n + 255) / 256, 256, 0, stream>>>(x, xbf, n);
    }
    {   // W_enc: K=512 (KS=16), N=1024 (NT=64), 1 matrix
        int total = 64 * 16 * 32 * 16;
        k_swizzle<<<(total + 255) / 256, 256, 0, stream>>>(
            W_enc, wenc_sw, 1024, 16, 64,
            (size_t)512 * 1024, (size_t)64 * 16 * 32 * 16, total);
    }
    {   // W_inner: 15 matrices, K=1024 (KS=32), N=1024 (NT=64)
        int total = N_INNER * 64 * 32 * 32 * 16;
        k_swizzle<<<(total + 255) / 256, 256, 0, stream>>>(
            W_inner, winner_sw, 1024, 32, 64,
            (size_t)1024 * 1024, (size_t)64 * 32 * 32 * 16, total);
    }
    {   // W_leaf: 16 matrices, K=1024 (KS=32), N=128 (NT=8)
        int total = N_LEAF * 8 * 32 * 32 * 16;
        k_swizzle<<<(total + 255) / 256, 256, 0, stream>>>(
            W_leaf, wleaf_sw, 128, 32, 8,
            (size_t)1024 * 128, (size_t)8 * 32 * 32 * 16, total);
    }

    // K1: encoder GEMM (+ReLU) -> bf16 emb
    k1_encoder<<<B_SZ / 16, 256, 0, stream>>>(xbf, wenc_sw, b_enc, emb);

    // K2: inner nodes, fused GEMM + argmax + sigmoid -> p_right (M=32 tiles)
    k2_inner<<<dim3(B_SZ / 32, N_INNER), 256, 0, stream>>>(
        emb, winner_sw, b_inner, x, p_right);

    // K3: leaf GEMMs + tree probs + weighted combine -> out
    k3_leaf<<<B_SZ / 16, 256, 0, stream>>>(emb, wleaf_sw, b_leaf, p_right, out);
}
